// PosUpdate_55628416418476
// MI455X (gfx1250) — matrix operand order repaired and checked by reading, not compiled, not run on hardware
//
#include <hip/hip_runtime.h>
#include <hip/hip_bf16.h>
#include <math.h>

// ---------------------------------------------------------------------------
// Problem constants (reference: B=8, N=256, H=128, IN_DIM=259)
// ---------------------------------------------------------------------------
#define Bn 8
#define Nn 256
#define Hn 128
#define INDIM 259
#define NWAVES 8
#define TPB 256

typedef __attribute__((ext_vector_type(16))) __bf16 v16bf;
typedef __attribute__((ext_vector_type(8)))  float  v8f;

// Workspace byte offsets
#define WS_WE    0            // bf16 [128*128]            32768 B
#define WS_W2    32768        // bf16 [128*128]            32768 B
#define WS_CB    65536        // f32  [128]                  512 B
#define WS_ROWA  66048        // f32  [B*N*128]          1048576 B
#define WS_COLB  1114624      // f32  [B*N*128]          1048576 B
// total 2163200 B (~2.06 MB) of d_ws used

// LDS byte offsets (dynamic shared) -- ~100KB => 3 workgroups per WGP (320KB)
#define S_WE    0             // bf16 [128*128]    32768
#define S_W2    32768         // bf16 [128*128]    32768
#define S_H1    65536         // bf16 [8][16*128]  32768
#define S_BIAS  98304         // f32  [128] rowA+cb  512
#define S_B2    98816         // f32  [128]          512
#define S_W3    99328         // f32  [128]          512
#define S_ACC   99840         // f32  [4]             16
#define SMEM_BYTES 99856

// ---------------------------------------------------------------------------
// Pre-kernel 1: fold edge path into We = W1c @ Wd (bf16), cb = b1 + W1c@bd,
// and convert W2 to bf16.  grid: 128 blocks x 128 threads.
// ---------------------------------------------------------------------------
__global__ void prep_weights(const float* __restrict__ W1,
                             const float* __restrict__ Wd,
                             const float* __restrict__ bd,
                             const float* __restrict__ b1,
                             const float* __restrict__ W2,
                             __bf16* __restrict__ wsWe,
                             __bf16* __restrict__ wsW2,
                             float*  __restrict__ wsCb) {
    const int h = blockIdx.x;   // output channel (row of We / W2)
    const int k = threadIdx.x;  // inner index
    float we = 0.f;
#pragma unroll
    for (int d = 0; d < 3; ++d)
        we += W1[h * INDIM + 256 + d] * Wd[d * Hn + k];
    wsWe[h * Hn + k] = (__bf16)we;
    wsW2[h * Hn + k] = (__bf16)W2[h * Hn + k];
    if (k == 0) {
        float cb = b1[h];
#pragma unroll
        for (int d = 0; d < 3; ++d)
            cb += W1[h * INDIM + 256 + d] * bd[d];
        wsCb[h] = cb;
    }
}

// ---------------------------------------------------------------------------
// Pre-kernel 2: rowA[bn,h] = W1[:, 0:128] @ x_emb[bn],
//               colB[bn,h] = W1[:, 128:256] @ x_emb[bn].
// grid: B*N blocks x 128 threads.
// ---------------------------------------------------------------------------
__global__ void prep_nodes(const float* __restrict__ x_emb,
                           const float* __restrict__ W1,
                           float* __restrict__ wsRowA,
                           float* __restrict__ wsColB) {
    const int bn = blockIdx.x;
    const int h  = threadIdx.x;
    __shared__ float xs[Hn];
    xs[h] = x_emb[(size_t)bn * Hn + h];
    __syncthreads();
    const float* w = W1 + (size_t)h * INDIM;
    float ra = 0.f, cb = 0.f;
#pragma unroll 8
    for (int k = 0; k < Hn; ++k) {
        ra += w[k]       * xs[k];
        cb += w[Hn + k]  * xs[k];
    }
    wsRowA[(size_t)bn * Hn + h] = ra;
    wsColB[(size_t)bn * Hn + h] = cb;
}

// ---------------------------------------------------------------------------
// B-fragment loader (two 16-byte chunks p[0..7], p[16..23] of a weight row -> 2x ds_load_b128)
// ---------------------------------------------------------------------------
__device__ __forceinline__ v16bf load_bfrag(const __bf16* __restrict__ p) {
    v16bf bf;
#pragma unroll
    for (int e = 0; e < 8; ++e) { bf[e] = p[e]; bf[8 + e] = p[16 + e]; }
    return bf;
}

// ---------------------------------------------------------------------------
// Main kernel: one workgroup (8 waves, wave32) per (b, i).
// Each wave handles j-tiles {wid, wid+8}: two bf16 WMMA GEMMs + silu + dot,
// then masked coord_diff accumulation over j.  B-fragments are software
// double-buffered so ds_load_b128 pipelines under the WMMAs.
// ---------------------------------------------------------------------------
__global__ void __launch_bounds__(TPB)
pos_update_main(const float* __restrict__ pair_emb,
                const float* __restrict__ pos,
                const float* __restrict__ coord_diff,
                const float* __restrict__ pair_mask,
                const float* __restrict__ b2,
                const float* __restrict__ W3,
                const __bf16* __restrict__ wsWe,
                const __bf16* __restrict__ wsW2,
                const float*  __restrict__ wsCb,
                const float*  __restrict__ wsRowA,
                const float*  __restrict__ wsColB,
                float* __restrict__ out) {
    extern __shared__ char smem[];
    __bf16* sWe   = (__bf16*)(smem + S_WE);
    __bf16* sW2   = (__bf16*)(smem + S_W2);
    __bf16* sH1   = (__bf16*)(smem + S_H1);
    float*  sBias = (float*) (smem + S_BIAS);
    float*  sB2   = (float*) (smem + S_B2);
    float*  sW3   = (float*) (smem + S_W3);
    float*  sAcc  = (float*) (smem + S_ACC);

    const int tid  = threadIdx.x;
    const int lane = tid & 31;
    const int wid  = tid >> 5;
    const int b = blockIdx.x / Nn;
    const int i = blockIdx.x % Nn;

    // ---- stage weights / per-(b,i) bias into LDS ----
    {
        const uint32_t* s1 = (const uint32_t*)wsWe;
        const uint32_t* s2 = (const uint32_t*)wsW2;
        uint32_t* d1 = (uint32_t*)sWe;
        uint32_t* d2 = (uint32_t*)sW2;
#pragma unroll
        for (int t = 0; t < 32; ++t) {
            d1[tid + TPB * t] = s1[tid + TPB * t];
            d2[tid + TPB * t] = s2[tid + TPB * t];
        }
        if (tid < Hn) {
            sBias[tid] = wsRowA[((size_t)b * Nn + i) * Hn + tid] + wsCb[tid];
            sB2[tid]   = b2[tid];
            sW3[tid]   = W3[tid];
        }
        if (tid < 4) sAcc[tid] = 0.f;
    }
    __syncthreads();

    // fragment lane geometry (per documented CDNA5 16-bit WMMA layouts)
    const int r   = lane & 15;          // A row within tile (pair index)
    const int kbA = (lane >> 4) * 8;    // A K-base split per half-wave
    const int kbB = (lane >> 4) * 8;    // B K-base split per half-wave
    const int nB  = lane & 15;          // B / C / D column
    const int mhi = (lane >> 4) * 8;    // C/D row offset per half-wave

    const v8f vzero = {0.f, 0.f, 0.f, 0.f, 0.f, 0.f, 0.f, 0.f};
    float tx = 0.f, ty = 0.f, tz = 0.f;
    const size_t rowBase = ((size_t)b * Nn + i) * Nn;
    const float* colBb = wsColB + (size_t)b * Nn * Hn;   // L2-resident (1MB)

    // per-lane bases into the weight matrices (column nB, K-half kbB)
    const __bf16* we_base = sWe + nB * Hn + kbB;
    const __bf16* w2_base = sW2 + nB * Hn + kbB;

    for (int jt = wid; jt < 16; jt += NWAVES) {
        const int j0 = jt * 16;
        const float* pe = pair_emb + (rowBase + j0) * Hn;

        if (jt + NWAVES < 16)
            __builtin_prefetch(pe + (size_t)NWAVES * 16 * Hn + r * Hn, 0, 1);

        // ---- GEMM1: pre1[16 pairs x 128] = pair_emb_tile x We^T ----
        v16bf A[4];
#pragma unroll
        for (int ks = 0; ks < 4; ++ks) {
            const float* p = pe + r * Hn + ks * 32 + kbA;
            v16bf a;
#pragma unroll
            for (int e = 0; e < 8; ++e) {
                a[e]     = (__bf16)p[e];
                a[8 + e] = (__bf16)p[16 + e];
            }
            A[ks] = a;
        }
        v8f acc[8];
#pragma unroll
        for (int nt = 0; nt < 8; ++nt) acc[nt] = vzero;
        {
            v16bf bcur = load_bfrag(we_base);
#pragma unroll
            for (int f = 0; f < 32; ++f) {          // f = nt*4 + ks
                v16bf bnext = bcur;
                if (f + 1 < 32) {
                    const int nt1 = (f + 1) >> 2, ks1 = (f + 1) & 3;
                    bnext = load_bfrag(we_base + nt1 * 16 * Hn + ks1 * 32);
                }
                const int nt = f >> 2, ks = f & 3;
                acc[nt] = __builtin_amdgcn_wmma_f32_16x16x32_bf16(
                    false, A[ks], false, bcur, (short)0, acc[nt], false, false);
                bcur = bnext;
            }
        }

        // ---- epilogue 1: silu(pre1 + bias + colB[j]) -> per-wave LDS ----
        __bf16* h1buf = sH1 + wid * (16 * Hn);
#pragma unroll
        for (int nt = 0; nt < 8; ++nt) {
            const int h = nt * 16 + nB;
            const float bias = sBias[h];
#pragma unroll
            for (int v = 0; v < 8; ++v) {
                const int M = v + mhi;
                float x = acc[nt][v] + bias + colBb[(size_t)(j0 + M) * Hn + h];
                float s1 = x * (1.f / (1.f + __expf(-x)));
                h1buf[M * Hn + h] = (__bf16)s1;
            }
        }
        // wave-local LDS RAW fence (CDNA5 split dep counter)
        asm volatile("s_wait_dscnt 0" ::: "memory");

        // ---- GEMM2: h2pre[16 x 128] = h1 x W2^T ----
#pragma unroll
        for (int ks = 0; ks < 4; ++ks) {
            const __bf16* p = h1buf + r * Hn + ks * 32 + kbA;
            v16bf a;
#pragma unroll
            for (int e = 0; e < 8; ++e) {
                a[e]     = p[e];
                a[8 + e] = p[16 + e];
            }
            A[ks] = a;
        }
        v8f acc2[8];
#pragma unroll
        for (int nt = 0; nt < 8; ++nt) acc2[nt] = vzero;
        {
            v16bf bcur = load_bfrag(w2_base);
#pragma unroll
            for (int f = 0; f < 32; ++f) {
                v16bf bnext = bcur;
                if (f + 1 < 32) {
                    const int nt1 = (f + 1) >> 2, ks1 = (f + 1) & 3;
                    bnext = load_bfrag(w2_base + nt1 * 16 * Hn + ks1 * 32);
                }
                const int nt = f >> 2, ks = f & 3;
                acc2[nt] = __builtin_amdgcn_wmma_f32_16x16x32_bf16(
                    false, A[ks], false, bcur, (short)0, acc2[nt], false, false);
                bcur = bnext;
            }
        }

        // ---- epilogue 2: h2 = silu(. + b2); s = W3 . h2 ; accumulate ----
        float sp[8];
#pragma unroll
        for (int v = 0; v < 8; ++v) sp[v] = 0.f;
#pragma unroll
        for (int nt = 0; nt < 8; ++nt) {
            const int h = nt * 16 + nB;
            const float w3h = sW3[h];
            const float b2h = sB2[h];
#pragma unroll
            for (int v = 0; v < 8; ++v) {
                float x = acc2[nt][v] + b2h;
                float h2 = x * (1.f / (1.f + __expf(-x)));
                sp[v] += w3h * h2;
            }
        }
#pragma unroll
        for (int v = 0; v < 8; ++v) {
            float s = sp[v];
            s += __shfl_xor(s, 1);
            s += __shfl_xor(s, 2);
            s += __shfl_xor(s, 4);
            s += __shfl_xor(s, 8);
            if (nB == 0) {                       // lanes 0 and 16: M = v + mhi
                const int j = j0 + v + mhi;
                const size_t pidx = rowBase + j;
                const float f = s * pair_mask[pidx];
                const float* cdp = coord_diff + pidx * 3;
                tx += cdp[0] * f;
                ty += cdp[1] * f;
                tz += cdp[2] * f;
            }
        }
    }

    // ---- workgroup reduction over j-tiles ----
    if (nB == 0) {
        atomicAdd(&sAcc[0], tx);
        atomicAdd(&sAcc[1], ty);
        atomicAdd(&sAcc[2], tz);
    }
    __syncthreads();
    if (tid < 3) {
        const size_t o = ((size_t)b * Nn + i) * 3 + tid;
        out[o] = pos[o] + sAcc[tid];
    }
}

// ---------------------------------------------------------------------------
// Launch
// ---------------------------------------------------------------------------
extern "C" void kernel_launch(void* const* d_in, const int* in_sizes, int n_in,
                              void* d_out, int out_size, void* d_ws, size_t ws_size,
                              hipStream_t stream) {
    const float* x_emb     = (const float*)d_in[0];
    const float* pair_emb  = (const float*)d_in[1];
    const float* pos       = (const float*)d_in[2];
    const float* coord_diff= (const float*)d_in[3];
    // d_in[4] node_mask: unused by the reference output
    const float* pair_mask = (const float*)d_in[5];
    const float* Wd        = (const float*)d_in[6];
    const float* bd        = (const float*)d_in[7];
    const float* W1        = (const float*)d_in[8];
    const float* b1        = (const float*)d_in[9];
    const float* W2        = (const float*)d_in[10];
    const float* b2        = (const float*)d_in[11];
    const float* W3        = (const float*)d_in[12];
    float* out = (float*)d_out;

    char* ws = (char*)d_ws;
    __bf16* wsWe  = (__bf16*)(ws + WS_WE);
    __bf16* wsW2  = (__bf16*)(ws + WS_W2);
    float*  wsCb  = (float*) (ws + WS_CB);
    float*  wsRowA= (float*) (ws + WS_ROWA);
    float*  wsColB= (float*) (ws + WS_COLB);

    (void)in_sizes; (void)n_in; (void)out_size; (void)ws_size;

    hipFuncSetAttribute((const void*)pos_update_main,
                        hipFuncAttributeMaxDynamicSharedMemorySize, SMEM_BYTES);

    prep_weights<<<Hn, Hn, 0, stream>>>(W1, Wd, bd, b1, W2, wsWe, wsW2, wsCb);
    prep_nodes<<<Bn * Nn, Hn, 0, stream>>>(x_emb, W1, wsRowA, wsColB);
    pos_update_main<<<Bn * Nn, TPB, SMEM_BYTES, stream>>>(
        pair_emb, pos, coord_diff, pair_mask, b2, W3,
        wsWe, wsW2, wsCb, wsRowA, wsColB, out);
}
